// PointMAE_73512660238516
// MI455X (gfx1250) — matrix operand order repaired and checked by reading, not compiled, not run on hardware
//
#include <hip/hip_runtime.h>
#include <hip/hip_bf16.h>
#include <math.h>

typedef __attribute__((ext_vector_type(16))) _Float16 v16h;
typedef __attribute__((ext_vector_type(2)))  _Float16 v2h;
typedef __attribute__((ext_vector_type(8)))  float    v8f;
typedef __attribute__((ext_vector_type(2)))  float    v2f;

#define BB   32
#define NPT  8192
#define GG   256
#define KNB  32
#define DD   384
#define HH   6
#define LLAY 12
#define FFN  1536
#define SEQ  64
#define HDIM 64
#define TOK  (BB*SEQ)   // 2048

#define WMMA_F16(a, b, c) \
  __builtin_amdgcn_wmma_f32_16x16x32_f16(false, (a), false, (b), (short)0, (c), false, false)

static __device__ __forceinline__ float gelu_f(float x) {
  float x3 = x * x * x;
  return 0.5f * x * (1.0f + tanhf(0.7978845608028654f * (x + 0.044715f * x3)));
}

// Fragment-major layouts (per 16x16x32 WMMA tile): [lane 0..31][e 0..15] contiguous (32B/lane).
// A tile (16x32, row m, K c):  lane = m + 16*((c&15)>>3), e = (c&7) + (c&16 ? 8 : 0)
// B tile (32x16, K kk, col n): lane = n + 16*((kk&15)>>3), e = (kk&7) + (kk&16 ? 8 : 0)

static __device__ __forceinline__ v16h frag_ld(const _Float16* p) {
  return *(const v16h*)p;
}

// ---------------- 0) weight swizzle: f32 (Kd x N) row-major -> f16 B-fragment blocks ----------------
// dst block (nt, ks) is 512 contiguous halves; flat: ((nt*(Kd/32)+ks)*32 + lane)*16 + e
// src element: gk = ks*32 + (lane>>4)*8 + (e&7) + (e>>3)*16 ; gn = nt*16 + (lane&15)

__global__ void pm_swizzle_w_kernel(const float* __restrict__ src, _Float16* __restrict__ dst,
                                    int Kd, int N) {
  int total = Kd * N;
  int ksTot = Kd >> 5;
  for (int i = blockIdx.x * blockDim.x + threadIdx.x; i < total; i += gridDim.x * blockDim.x) {
    int e = i & 15;
    int lane = (i >> 4) & 31;
    int blk = i >> 9;
    int ks = blk % ksTot;
    int nt = blk / ksTot;
    int gk = ks * 32 + ((lane >> 4) << 3) + (e & 7) + ((e >> 3) << 4);
    int gn = nt * 16 + (lane & 15);
    dst[i] = (_Float16)src[(size_t)gk * N + gn];
  }
}

// ---------------- 1) normalize points per batch ----------------

__global__ void pm_normalize_kernel(const float* __restrict__ pts, float* __restrict__ xn) {
  int b = blockIdx.x, tid = threadIdx.x; // 256 threads
  __shared__ float rA[3][256];
  __shared__ float rB[3][256];
  __shared__ float center[3];
  __shared__ float invrad;
  const float* p = pts + (size_t)b * NPT * 3;
  float mn[3] = {1e30f, 1e30f, 1e30f}, mx[3] = {-1e30f, -1e30f, -1e30f};
  for (int i = tid; i < NPT; i += 256) {
#pragma unroll
    for (int c = 0; c < 3; ++c) {
      float v = p[i * 3 + c];
      mn[c] = fminf(mn[c], v);
      mx[c] = fmaxf(mx[c], v);
    }
  }
#pragma unroll
  for (int c = 0; c < 3; ++c) { rA[c][tid] = mn[c]; rB[c][tid] = mx[c]; }
  __syncthreads();
  for (int o = 128; o > 0; o >>= 1) {
    if (tid < o) {
#pragma unroll
      for (int c = 0; c < 3; ++c) {
        rA[c][tid] = fminf(rA[c][tid], rA[c][tid + o]);
        rB[c][tid] = fmaxf(rB[c][tid], rB[c][tid + o]);
      }
    }
    __syncthreads();
  }
  if (tid == 0) {
#pragma unroll
    for (int c = 0; c < 3; ++c) center[c] = 0.5f * (rA[c][0] + rB[c][0]);
  }
  __syncthreads();
  float mr = 0.f;
  float cx = center[0], cy = center[1], cz = center[2];
  for (int i = tid; i < NPT; i += 256) {
    float dx = p[i * 3 + 0] - cx, dy = p[i * 3 + 1] - cy, dz = p[i * 3 + 2] - cz;
    mr = fmaxf(mr, dx * dx + dy * dy + dz * dz);
  }
  __syncthreads();
  rA[0][tid] = mr;
  __syncthreads();
  for (int o = 128; o > 0; o >>= 1) {
    if (tid < o) rA[0][tid] = fmaxf(rA[0][tid], rA[0][tid + o]);
    __syncthreads();
  }
  if (tid == 0) invrad = 1.0f / sqrtf(rA[0][0]);
  __syncthreads();
  float ir = invrad;
  float* xo = xn + (size_t)b * NPT * 3;
  for (int i = tid; i < NPT; i += 256) {
    xo[i * 3 + 0] = (p[i * 3 + 0] - cx) * ir;
    xo[i * 3 + 1] = (p[i * 3 + 1] - cy) * ir;
    xo[i * 3 + 2] = (p[i * 3 + 2] - cz) * ir;
  }
}

// ---------------- 2) farthest point sampling ----------------

__global__ void pm_fps_kernel(const float* __restrict__ X, int* __restrict__ cidx,
                              float* __restrict__ centers) {
  int b = blockIdx.x, tid = threadIdx.x; // 256 threads
  __shared__ float dists[NPT];
  __shared__ float rv[256];
  __shared__ int   ri[256];
  __shared__ float lastp[3];
  const float* xb = X + (size_t)b * NPT * 3;
  for (int i = tid; i < NPT; i += 256) dists[i] = 1e10f;
  if (tid == 0) {
    cidx[b * GG + 0] = 0;
    centers[(size_t)(b * GG + 0) * 3 + 0] = xb[0];
    centers[(size_t)(b * GG + 0) * 3 + 1] = xb[1];
    centers[(size_t)(b * GG + 0) * 3 + 2] = xb[2];
    lastp[0] = xb[0]; lastp[1] = xb[1]; lastp[2] = xb[2];
  }
  __syncthreads();
  for (int it = 1; it < GG; ++it) {
    float lx = lastp[0], ly = lastp[1], lz = lastp[2];
    float bestv = -1.f;
    int besti = NPT;
    for (int i = tid; i < NPT; i += 256) {
      float dx = xb[i * 3 + 0] - lx, dy = xb[i * 3 + 1] - ly, dz = xb[i * 3 + 2] - lz;
      float d = dx * dx + dy * dy + dz * dz;
      float dd = fminf(dists[i], d);
      dists[i] = dd;
      if (dd > bestv) { bestv = dd; besti = i; }
    }
    rv[tid] = bestv; ri[tid] = besti;
    __syncthreads();
    for (int o = 128; o > 0; o >>= 1) {
      if (tid < o) {
        float v2 = rv[tid + o]; int i2 = ri[tid + o];
        if (v2 > rv[tid] || (v2 == rv[tid] && i2 < ri[tid])) { rv[tid] = v2; ri[tid] = i2; }
      }
      __syncthreads();
    }
    if (tid == 0) {
      int nx = ri[0];
      cidx[b * GG + it] = nx;
      float px = xb[nx * 3 + 0], py = xb[nx * 3 + 1], pz = xb[nx * 3 + 2];
      centers[(size_t)(b * GG + it) * 3 + 0] = px;
      centers[(size_t)(b * GG + it) * 3 + 1] = py;
      centers[(size_t)(b * GG + it) * 3 + 2] = pz;
      lastp[0] = px; lastp[1] = py; lastp[2] = pz;
    }
    __syncthreads();
  }
}

// ---------------- 3) kNN grouping (K smallest d2, index tie-break) ----------------

__global__ void pm_knn_kernel(const float* __restrict__ X, const float* __restrict__ centers,
                              float* __restrict__ patches) {
  int g = blockIdx.x, b = blockIdx.y, tid = threadIdx.x; // 256 threads
  __shared__ float d2[NPT];
  __shared__ float rv[256];
  __shared__ int   ri[256];
  const float* xb = X + (size_t)b * NPT * 3;
  const float* c = centers + (size_t)(b * GG + g) * 3;
  float cx = c[0], cy = c[1], cz = c[2];
  for (int i = tid; i < NPT; i += 256) {
    float dx = xb[i * 3 + 0] - cx, dy = xb[i * 3 + 1] - cy, dz = xb[i * 3 + 2] - cz;
    d2[i] = dx * dx + dy * dy + dz * dz;
  }
  __syncthreads();
  for (int k = 0; k < KNB; ++k) {
    float bestv = 1e30f;
    int besti = NPT;
    for (int i = tid; i < NPT; i += 256) {
      float v = d2[i];
      if (v < bestv) { bestv = v; besti = i; }
    }
    rv[tid] = bestv; ri[tid] = besti;
    __syncthreads();
    for (int o = 128; o > 0; o >>= 1) {
      if (tid < o) {
        float v2 = rv[tid + o]; int i2 = ri[tid + o];
        if (v2 < rv[tid] || (v2 == rv[tid] && i2 < ri[tid])) { rv[tid] = v2; ri[tid] = i2; }
      }
      __syncthreads();
    }
    if (tid == 0) {
      int ix = ri[0];
      size_t op = ((size_t)((b * GG + g) * KNB + k)) * 3;
      patches[op + 0] = xb[ix * 3 + 0] - cx;
      patches[op + 1] = xb[ix * 3 + 1] - cy;
      patches[op + 2] = xb[ix * 3 + 2] - cz;
      d2[ix] = 1e30f;
    }
    __syncthreads();
  }
}

// ---------------- 4) fused patch MLP: 3->128 relu -> WMMA 128->256 -> max over K ----------------
// W2f: pre-swizzled B-fragment blocks (Kd=128 -> ksTot=4, 16 ntiles)

__global__ void pm_patch_mlp_kernel(const float* __restrict__ patches,
                                    const float* __restrict__ W1, const float* __restrict__ b1,
                                    const _Float16* __restrict__ W2f, const float* __restrict__ b2,
                                    float* __restrict__ tok256) {
  int g = blockIdx.x, b = blockIdx.y, tid = threadIdx.x; // 256 threads, 8 waves
  int lane = tid & 31, wv = tid >> 5;
  __shared__ float xs[KNB][3];
  __shared__ alignas(32) _Float16 F1F[2][4][32][16];  // [mt][kstep][lane][e]  A-layout
  __shared__ float Out[KNB][256];
  const float* pp = patches + ((size_t)(b * GG + g)) * KNB * 3;
  for (int i = tid; i < KNB * 3; i += 256) xs[i / 3][i % 3] = pp[i];
  __syncthreads();
  // stage1: pairs (p, j even): 32*64 = 2048 tasks
  for (int i = tid; i < 2048; i += 256) {
    int p = i >> 6, jp = i & 63, j = jp * 2;
    float x0 = xs[p][0], x1 = xs[p][1], x2 = xs[p][2];
    float a0 = b1[j]     + x0 * W1[j]     + x1 * W1[128 + j]     + x2 * W1[256 + j];
    float a1 = b1[j + 1] + x0 * W1[j + 1] + x1 * W1[128 + j + 1] + x2 * W1[256 + j + 1];
    int c = j & 31, ks = j >> 5;
    int ha = (c & 15) >> 3, ea = (c & 7) + ((c & 16) ? 8 : 0);
    v2h t; t[0] = (_Float16)fmaxf(a0, 0.f); t[1] = (_Float16)fmaxf(a1, 0.f);
    *(v2h*)&F1F[p >> 4][ks][(p & 15) + 16 * ha][ea] = t;
  }
  __syncthreads();
  int mt = wv & 1;
  int ntb = (wv >> 1) * 4;
  v8f z = {0.f, 0.f, 0.f, 0.f, 0.f, 0.f, 0.f, 0.f};
  v8f acc0 = z, acc1 = z, acc2 = z, acc3 = z;
#pragma unroll
  for (int ks = 0; ks < 4; ++ks) {
    v16h af = frag_ld(&F1F[mt][ks][lane][0]);
    const _Float16* wb = W2f + ((((size_t)ntb * 4) + ks) << 9) + (lane << 4);
    acc0 = WMMA_F16(af, frag_ld(wb),                   acc0);
    acc1 = WMMA_F16(af, frag_ld(wb + (4u << 9)),       acc1);
    acc2 = WMMA_F16(af, frag_ld(wb + (8u << 9)),       acc2);
    acc3 = WMMA_F16(af, frag_ld(wb + (12u << 9)),      acc3);
  }
  int cl = lane & 15, half = lane >> 4;
#pragma unroll
  for (int t = 0; t < 4; ++t) {
    v8f a = (t == 0) ? acc0 : (t == 1) ? acc1 : (t == 2) ? acc2 : acc3;
    int col = (ntb + t) * 16 + cl;
    float bc = b2[col];
#pragma unroll
    for (int r = 0; r < 8; ++r) Out[mt * 16 + r + 8 * half][col] = a[r] + bc;
  }
  __syncthreads();
  float m = Out[0][tid];
#pragma unroll
  for (int r = 1; r < KNB; ++r) m = fmaxf(m, Out[r][tid]);
  tok256[((size_t)(b * GG + g)) * 256 + tid] = m;
}

// ---------------- 5) pos MLP stage1: centers (3) -> 128 relu ----------------

__global__ void pm_pos1_kernel(const float* __restrict__ centers, const float* __restrict__ w1,
                               const float* __restrict__ b1, float* __restrict__ out) {
  int total = BB * GG * 128;
  for (int i = blockIdx.x * blockDim.x + threadIdx.x; i < total; i += gridDim.x * blockDim.x) {
    int r = i >> 7, j = i & 127;
    const float* c = centers + (size_t)r * 3;
    float a = b1[j] + c[0] * w1[j] + c[1] * w1[128 + j] + c[2] * w1[256 + j];
    out[i] = fmaxf(a, 0.f);
  }
}

// ---------------- 6) masking: stable argsort rank, keep 64 smallest-noise ids ----------------

__global__ void pm_keep_kernel(const float* __restrict__ noise, int* __restrict__ keep) {
  int b = blockIdx.x, i = threadIdx.x; // 256 threads
  float v = noise[b * GG + i];
  int rank = 0;
  for (int j = 0; j < GG; ++j) {
    float vj = noise[b * GG + j];
    rank += (vj < v) || (vj == v && j < i);
  }
  if (rank < SEQ) keep[b * SEQ + rank] = i;
}

__global__ void pm_gather_kernel(const float* __restrict__ tok, const float* __restrict__ pos,
                                 const int* __restrict__ keep, float* __restrict__ h) {
  int total = TOK * DD;
  for (int i = blockIdx.x * blockDim.x + threadIdx.x; i < total; i += gridDim.x * blockDim.x) {
    int row = i / DD, d = i % DD;
    int b = row >> 6, s = row & 63;
    int g = keep[b * SEQ + s];
    size_t src = ((size_t)(b * GG + g)) * DD + d;
    h[i] = tok[src] + pos[src];
  }
}

// ---------------- 7) generic WMMA GEMM: C = act(A@Wf + bias) (+ residual) ----------------
// A: (M,Kd) f32; Wf: pre-swizzled f16 B-fragment blocks; 64x64 tile, BK=32, 256 thr / 8 waves.
// B fragments are loaded directly from global (coalesced 32B/lane), A staged via LDS.
// Requires M%64==0, N%64==0, Kd%32==0 (true for all call sites).

template <int ACT>
__global__ void pm_gemm_kernel(const float* __restrict__ A, const _Float16* __restrict__ Wf,
                               const float* __restrict__ bias, const float* __restrict__ residual,
                               float* __restrict__ C, int M, int N, int Kd) {
  __shared__ alignas(32) _Float16 AsF[4][32][16];  // [mtile][lane][e]  A-layout
  int tid = threadIdx.x;
  int lane = tid & 31, wv = tid >> 5;
  int m0 = blockIdx.y * 64, n0 = blockIdx.x * 64;
  int mt = wv & 3;
  int ntb = (wv >> 2) * 2;
  int ksTot = Kd >> 5;
  // per-wave base for B fragment stream: block (nt, ks) at ((nt*ksTot + ks)*512
  const _Float16* wb0 = Wf + (((size_t)((n0 >> 4) + ntb) * ksTot) << 9) + (lane << 4);
  const _Float16* wb1 = wb0 + ((size_t)ksTot << 9);
  v8f z = {0.f, 0.f, 0.f, 0.f, 0.f, 0.f, 0.f, 0.f};
  v8f acc0 = z, acc1 = z;
  for (int ks = 0; ks < ksTot; ++ks) {
    int k0 = ks << 5;
    // A tile: 4 mtiles * 32 lanes * 8 e-pairs = 1024 packed pairs
    for (int i = tid; i < 1024; i += 256) {
      int ep = i & 7, ln = (i >> 3) & 31, mtile = i >> 8;
      int e = ep * 2, half = ln >> 4;
      int c = (e & 7) + half * 8 + ((e >> 3) << 4);  // inverse e-map; even
      int gm = m0 + mtile * 16 + (ln & 15);
      v2f av = *(const v2f*)&A[(size_t)gm * Kd + k0 + c];
      v2h t; t[0] = (_Float16)av.x; t[1] = (_Float16)av.y;
      *(v2h*)&AsF[mtile][ln][e] = t;
    }
    if (k0 + 32 < Kd)
      __builtin_prefetch(&A[(size_t)(m0 + (tid & 63)) * Kd + k0 + 32], 0, 1);
    __syncthreads();
    v16h af = frag_ld(&AsF[mt][lane][0]);
    acc0 = WMMA_F16(af, frag_ld(wb0 + ((size_t)ks << 9)), acc0);
    acc1 = WMMA_F16(af, frag_ld(wb1 + ((size_t)ks << 9)), acc1);
    __syncthreads();
  }
  int cl = lane & 15, half = lane >> 4;
#pragma unroll
  for (int t = 0; t < 2; ++t) {
    v8f a = (t == 0) ? acc0 : acc1;
    int gcol = n0 + (ntb + t) * 16 + cl;
    float bc = bias ? bias[gcol] : 0.f;
#pragma unroll
    for (int r = 0; r < 8; ++r) {
      int grow = m0 + mt * 16 + r + 8 * half;
      float v = a[r] + bc;
      if (ACT == 1) v = fmaxf(v, 0.f);
      if (ACT == 2) v = gelu_f(v);
      if (residual) v += residual[(size_t)grow * N + gcol];
      C[(size_t)grow * N + gcol] = v;
    }
  }
}

// ---------------- 8) layernorm ----------------

__global__ void pm_layernorm_kernel(const float* __restrict__ X, const float* __restrict__ gb,
                                    float* __restrict__ Y) {
  int row = blockIdx.x, tid = threadIdx.x; // 128 threads
  const float* x = X + (size_t)row * DD;
  float* y = Y + (size_t)row * DD;
  __shared__ float red[128];
  float s = 0.f;
  for (int i = tid; i < DD; i += 128) s += x[i];
  red[tid] = s;
  __syncthreads();
  for (int o = 64; o > 0; o >>= 1) {
    if (tid < o) red[tid] += red[tid + o];
    __syncthreads();
  }
  float mean = red[0] * (1.0f / DD);
  __syncthreads();
  float v = 0.f;
  for (int i = tid; i < DD; i += 128) {
    float d = x[i] - mean;
    v += d * d;
  }
  red[tid] = v;
  __syncthreads();
  for (int o = 64; o > 0; o >>= 1) {
    if (tid < o) red[tid] += red[tid + o];
    __syncthreads();
  }
  float inv = rsqrtf(red[0] * (1.0f / DD) + 1e-5f);
  for (int i = tid; i < DD; i += 128) y[i] = (x[i] - mean) * inv * gb[i] + gb[DD + i];
}

// ---------------- 9) attention per (b,h): S=64, HD=64 ----------------

__global__ void pm_attention_kernel(const float* __restrict__ qkv, float* __restrict__ O) {
  int hh = blockIdx.x, b = blockIdx.y; // grid (H, B), 256 threads
  int tid = threadIdx.x, lane = tid & 31, wv = tid >> 5;
  __shared__ alignas(32) _Float16 QF[4][2][32][16]; // [mt][ks][lane][e]  A-layout of Q (K=d)
  __shared__ alignas(32) _Float16 KF[4][2][32][16]; // [nt][ks][lane][e]  B-layout of K^T (kk=d, n=t)
  __shared__ alignas(32) _Float16 VF[4][2][32][16]; // [nt][ks][lane][e]  B-layout of V (kk=t, n=d)
  __shared__ alignas(32) _Float16 PF[4][2][32][16]; // [mt][ks][lane][e]  A-layout of P (K=t)
  __shared__ float Ss[64][68];
  // Q,K: iterate (token s, d-pair)
  for (int i = tid; i < 2048; i += 256) {
    int s = i >> 5, dp = i & 31, d = dp * 2;
    size_t base = ((size_t)(b * SEQ + s)) * (3 * DD) + hh * HDIM + d;
    v2f q = *(const v2f*)&qkv[base];
    v2f k = *(const v2f*)&qkv[base + DD];
    int c = d & 31, ks = d >> 5;
    int ha = (c & 15) >> 3, ea = (c & 7) + ((c & 16) ? 8 : 0);
    v2h tq; tq[0] = (_Float16)q.x; tq[1] = (_Float16)q.y;
    *(v2h*)&QF[s >> 4][ks][(s & 15) + 16 * ha][ea] = tq;
    int hb = (d >> 3) & 1, eb = (d & 7) + ((d & 16) ? 8 : 0);
    v2h tk; tk[0] = (_Float16)k.x; tk[1] = (_Float16)k.y;
    *(v2h*)&KF[s >> 4][ks][(s & 15) + 16 * hb][eb] = tk;
  }
  // V: iterate (d, token-pair) since n = d for V's B-layout
  for (int i = tid; i < 2048; i += 256) {
    int d = i >> 5, sp = i & 31, s = sp * 2;
    size_t base = ((size_t)(b * SEQ + s)) * (3 * DD) + hh * HDIM + 2 * DD + d;
    float v0 = qkv[base], v1 = qkv[base + 3 * DD];
    int ks = s >> 5, hb = (s >> 3) & 1, eb = (s & 7) + ((s & 16) ? 8 : 0);
    v2h tv; tv[0] = (_Float16)v0; tv[1] = (_Float16)v1;
    *(v2h*)&VF[d >> 4][ks][(d & 15) + 16 * hb][eb] = tv;
  }
  __syncthreads();
  int mt = wv & 3, ntb = (wv >> 2) * 2;
  v8f z = {0.f, 0.f, 0.f, 0.f, 0.f, 0.f, 0.f, 0.f};
  v8f s0 = z, s1 = z;
#pragma unroll
  for (int ks = 0; ks < 2; ++ks) {
    v16h aq = frag_ld(&QF[mt][ks][lane][0]);
    s0 = WMMA_F16(aq, frag_ld(&KF[ntb + 0][ks][lane][0]), s0);
    s1 = WMMA_F16(aq, frag_ld(&KF[ntb + 1][ks][lane][0]), s1);
  }
  int cl = lane & 15, half = lane >> 4;
#pragma unroll
  for (int t = 0; t < 2; ++t) {
    v8f a = (t == 0) ? s0 : s1;
    int col = (ntb + t) * 16 + cl;
#pragma unroll
    for (int r = 0; r < 8; ++r) Ss[mt * 16 + r + 8 * half][col] = a[r] * 0.125f; // 1/sqrt(64)
  }
  __syncthreads();
  if (tid < 64) {
    float mx = -1e30f;
    for (int j = 0; j < 64; ++j) mx = fmaxf(mx, Ss[tid][j]);
    float sum = 0.f;
    for (int j = 0; j < 64; ++j) {
      float e = expf(Ss[tid][j] - mx);
      Ss[tid][j] = e;
      sum += e;
    }
    float inv = 1.0f / sum;
    for (int j = 0; j < 64; ++j) Ss[tid][j] *= inv;
  }
  __syncthreads();
  // P -> A-layout (K dim = key token t), packed pairs over even t
  for (int i = tid; i < 2048; i += 256) {
    int s = i >> 5, tp = i & 31, t = tp * 2;
    int c = t & 31, ks = t >> 5;
    int ha = (c & 15) >> 3, ea = (c & 7) + ((c & 16) ? 8 : 0);
    v2h tpk; tpk[0] = (_Float16)Ss[s][t]; tpk[1] = (_Float16)Ss[s][t + 1];
    *(v2h*)&PF[s >> 4][ks][(s & 15) + 16 * ha][ea] = tpk;
  }
  __syncthreads();
  v8f o0 = z, o1 = z;
#pragma unroll
  for (int ks = 0; ks < 2; ++ks) {
    v16h ap = frag_ld(&PF[mt][ks][lane][0]);
    o0 = WMMA_F16(ap, frag_ld(&VF[ntb + 0][ks][lane][0]), o0);
    o1 = WMMA_F16(ap, frag_ld(&VF[ntb + 1][ks][lane][0]), o1);
  }
#pragma unroll
  for (int t = 0; t < 2; ++t) {
    v8f a = (t == 0) ? o0 : o1;
    int d = (ntb + t) * 16 + cl;
#pragma unroll
    for (int r = 0; r < 8; ++r) {
      int s = mt * 16 + r + 8 * half;
      O[((size_t)(b * SEQ + s)) * DD + hh * HDIM + d] = a[r];
    }
  }
}

// ---------------- host launch ----------------

extern "C" void kernel_launch(void* const* d_in, const int* in_sizes, int n_in,
                              void* d_out, int out_size, void* d_ws, size_t ws_size,
                              hipStream_t stream) {
  (void)in_sizes; (void)n_in; (void)out_size; (void)ws_size;
  const float* points = (const float*)d_in[0];
  const float* noise  = (const float*)d_in[1];
  const float* pe_w1  = (const float*)d_in[2];
  const float* pe_b1  = (const float*)d_in[3];
  const float* pe_w2  = (const float*)d_in[4];
  const float* pe_b2  = (const float*)d_in[5];
  const float* pe_w3  = (const float*)d_in[6];
  const float* pe_b3  = (const float*)d_in[7];
  const float* pos_w1 = (const float*)d_in[8];
  const float* pos_b1 = (const float*)d_in[9];
  const float* pos_w2 = (const float*)d_in[10];
  const float* pos_b2 = (const float*)d_in[11];
  const float* ln1    = (const float*)d_in[12];
  const float* wqkv   = (const float*)d_in[13];
  const float* bqkv   = (const float*)d_in[14];
  const float* wo     = (const float*)d_in[15];
  const float* bo     = (const float*)d_in[16];
  const float* ln2    = (const float*)d_in[17];
  const float* wi     = (const float*)d_in[18];
  const float* bi     = (const float*)d_in[19];
  const float* wo2    = (const float*)d_in[20];
  const float* bo2    = (const float*)d_in[21];
  const float* lnf    = (const float*)d_in[22];

  char* ws = (char*)d_ws;
  size_t off = 0;
  auto carve = [&](size_t bytes) -> void* {
    void* p = ws + off;
    off += (bytes + 255) & ~(size_t)255;
    return p;
  };
  float* xn      = (float*)carve((size_t)BB * NPT * 3 * 4);
  int*   cidx    = (int*)  carve((size_t)BB * GG * 4);
  float* centers = (float*)carve((size_t)BB * GG * 3 * 4);
  float* patches = (float*)carve((size_t)BB * GG * KNB * 3 * 4);
  float* tok256  = (float*)carve((size_t)BB * GG * 256 * 4);
  float* tok384  = (float*)carve((size_t)BB * GG * DD * 4);
  float* posh    = (float*)carve((size_t)BB * GG * 128 * 4);
  float* pos384  = (float*)carve((size_t)BB * GG * DD * 4);
  int*   keep    = (int*)  carve((size_t)BB * SEQ * 4);
  float* hbuf    = (float*)carve((size_t)TOK * DD * 4);
  float* ybuf    = (float*)carve((size_t)TOK * DD * 4);
  float* qkvbuf  = (float*)carve((size_t)TOK * 3 * DD * 4);
  float* obuf    = (float*)carve((size_t)TOK * DD * 4);
  float* ffnbuf  = (float*)carve((size_t)TOK * FFN * 4);
  // f16 fragment-swizzled weight mirrors
  const int n_pe_w2  = 128 * 256;
  const int n_pe_w3  = 256 * DD;
  const int n_pos_w2 = 128 * DD;
  const int n_wqkv1  = DD * 3 * DD;
  const int n_wo1    = DD * DD;
  const int n_wi1    = DD * FFN;
  const int n_wo21   = FFN * DD;
  _Float16* pe_w2h  = (_Float16*)carve((size_t)n_pe_w2 * 2);
  _Float16* pe_w3h  = (_Float16*)carve((size_t)n_pe_w3 * 2);
  _Float16* pos_w2h = (_Float16*)carve((size_t)n_pos_w2 * 2);
  _Float16* wqkvh   = (_Float16*)carve((size_t)n_wqkv1 * LLAY * 2);
  _Float16* woh     = (_Float16*)carve((size_t)n_wo1 * LLAY * 2);
  _Float16* wih     = (_Float16*)carve((size_t)n_wi1 * LLAY * 2);
  _Float16* wo2h    = (_Float16*)carve((size_t)n_wo21 * LLAY * 2);

  // weight swizzle+convert (f32 row-major -> f16 B-fragment blocks)
  pm_swizzle_w_kernel<<<512, 256, 0, stream>>>(pe_w2, pe_w2h, 128, 256);
  pm_swizzle_w_kernel<<<512, 256, 0, stream>>>(pe_w3, pe_w3h, 256, DD);
  pm_swizzle_w_kernel<<<512, 256, 0, stream>>>(pos_w2, pos_w2h, 128, DD);
  for (int l = 0; l < LLAY; ++l) {
    pm_swizzle_w_kernel<<<1024, 256, 0, stream>>>(wqkv + (size_t)l * n_wqkv1,
                                                  wqkvh + (size_t)l * n_wqkv1, DD, 3 * DD);
    pm_swizzle_w_kernel<<<512, 256, 0, stream>>>(wo + (size_t)l * n_wo1,
                                                 woh + (size_t)l * n_wo1, DD, DD);
    pm_swizzle_w_kernel<<<1024, 256, 0, stream>>>(wi + (size_t)l * n_wi1,
                                                  wih + (size_t)l * n_wi1, DD, FFN);
    pm_swizzle_w_kernel<<<1024, 256, 0, stream>>>(wo2 + (size_t)l * n_wo21,
                                                  wo2h + (size_t)l * n_wo21, FFN, DD);
  }

  // geometry pipeline
  pm_normalize_kernel<<<BB, 256, 0, stream>>>(points, xn);
  pm_fps_kernel<<<BB, 256, 0, stream>>>(xn, cidx, centers);
  pm_knn_kernel<<<dim3(GG, BB), 256, 0, stream>>>(xn, centers, patches);
  pm_patch_mlp_kernel<<<dim3(GG, BB), 256, 0, stream>>>(patches, pe_w1, pe_b1, pe_w2h, pe_b2, tok256);
  // tokens: (B*G,256) @ (256,384) + b3
  pm_gemm_kernel<0><<<dim3(DD / 64, (BB * GG) / 64), 256, 0, stream>>>(
      tok256, pe_w3h, pe_b3, nullptr, tok384, BB * GG, DD, 256);
  // pos embed
  pm_pos1_kernel<<<1024, 256, 0, stream>>>(centers, pos_w1, pos_b1, posh);
  pm_gemm_kernel<0><<<dim3(DD / 64, (BB * GG) / 64), 256, 0, stream>>>(
      posh, pos_w2h, pos_b2, nullptr, pos384, BB * GG, DD, 128);
  // masking + gather
  pm_keep_kernel<<<BB, GG, 0, stream>>>(noise, keep);
  pm_gather_kernel<<<1024, 256, 0, stream>>>(tok384, pos384, keep, hbuf);

  // transformer
  for (int l = 0; l < LLAY; ++l) {
    const float* ln1p = ln1 + (size_t)l * 2 * DD;
    const float* ln2p = ln2 + (size_t)l * 2 * DD;
    pm_layernorm_kernel<<<TOK, 128, 0, stream>>>(hbuf, ln1p, ybuf);
    pm_gemm_kernel<0><<<dim3((3 * DD) / 64, TOK / 64), 256, 0, stream>>>(
        ybuf, wqkvh + (size_t)l * n_wqkv1, bqkv + (size_t)l * 3 * DD, nullptr,
        qkvbuf, TOK, 3 * DD, DD);
    pm_attention_kernel<<<dim3(HH, BB), 256, 0, stream>>>(qkvbuf, obuf);
    pm_gemm_kernel<0><<<dim3(DD / 64, TOK / 64), 256, 0, stream>>>(
        obuf, woh + (size_t)l * n_wo1, bo + (size_t)l * DD, hbuf, hbuf, TOK, DD, DD);
    pm_layernorm_kernel<<<TOK, 128, 0, stream>>>(hbuf, ln2p, ybuf);
    pm_gemm_kernel<2><<<dim3(FFN / 64, TOK / 64), 256, 0, stream>>>(
        ybuf, wih + (size_t)l * n_wi1, bi + (size_t)l * FFN, nullptr, ffnbuf, TOK, FFN, DD);
    pm_gemm_kernel<0><<<dim3(DD / 64, TOK / 64), 256, 0, stream>>>(
        ffnbuf, wo2h + (size_t)l * n_wo21, bo2 + (size_t)l * DD, hbuf, hbuf, TOK, DD, FFN);
  }
  pm_layernorm_kernel<<<TOK, 128, 0, stream>>>(hbuf, lnf, (float*)d_out);
}
